// EdgeGCNV2_58944131170464
// MI455X (gfx1250) — compile-verified, matrix operand order checked by reading, not run on hardware
//
#include <hip/hip_runtime.h>
#include <cstdint>
#include <cstddef>

// ---------------- CDNA5 WMMA types ----------------
typedef __attribute__((ext_vector_type(16))) __bf16 v16bf;
typedef __attribute__((ext_vector_type(8)))  __bf16 v8bf;
typedef __attribute__((ext_vector_type(4)))  __bf16 v4bf;
typedef __attribute__((ext_vector_type(8)))  float  v8f;

#define LDT 40   // padded element stride for [outer][32] LDS tiles (80 B rows, 16-B aligned)
#define LDW 136  // padded element stride for [outer][128] LDS tiles (272 B rows, 16-B aligned)

__device__ inline v8f wmma_bf16(v16bf a, v16bf b, v8f c) {
  // D = A(16x32 bf16) x B(32x16 bf16) + C(16x16 f32)
  return __builtin_amdgcn_wmma_f32_16x16x32_bf16(
      /*neg_a=*/false, a, /*neg_b=*/false, b,
      /*c_mod=*/(short)0, c, /*reuse_a=*/false, /*reuse_b=*/false);
}

// Unified fragment loader. LDS tile is [outer][k] row-major with element stride ld.
// For A: outer = matrix row. For B (stored transposed): outer = matrix column.
// Lane l<16: outer=obase+l, K = kc+{0..7,16..23}; lane>=16: outer=obase+l-16, K = kc+{8..15,24..31}.
// Each lane's 16 elements = two contiguous 16-byte runs -> two ds_load_b128.
__device__ inline v16bf load_frag(const __bf16* base, int ld, int obase, int kc, int lane) {
  const int o  = obase + (lane & 15);
  const int kb = (lane < 16) ? 0 : 8;
  const __bf16* p = base + (size_t)o * ld + kc + kb;
  const v8bf lo = *reinterpret_cast<const v8bf*>(p);
  const v8bf hi = *reinterpret_cast<const v8bf*>(p + 16);
  return __builtin_shufflevector(lo, hi, 0, 1, 2, 3, 4, 5, 6, 7,
                                 8, 9, 10, 11, 12, 13, 14, 15);
}

// ---------------- init: zero accumulators, -inf maxes, transposed bf16 weights ----------------
__global__ void init_kernel(float* m, float* denom, float* gat, int N, int gatSz,
                            const float* Wg, __bf16* WgBT,
                            const float* We, __bf16* WeBT, int szW,
                            const float* Wl, __bf16* WlBT, int szWl) {
  const int stride = gridDim.x * blockDim.x;
  const int tid = blockIdx.x * blockDim.x + threadIdx.x;
  for (int k = tid; k < gatSz; k += stride) gat[k] = 0.0f;
  for (int k = tid; k < N; k += stride) {
    m[k] = __uint_as_float(0xff800000u);  // -inf
    denom[k] = 0.0f;
  }
  for (int k = tid; k < szW; k += stride) {  // [128][128] -> [n][k]
    const int r = k >> 7, c = k & 127;
    WgBT[c * 128 + r] = (__bf16)Wg[k];
    WeBT[c * 128 + r] = (__bf16)We[k];
  }
  for (int k = tid; k < szWl; k += stride) { // [128][64] -> [n=64][k=128]
    const int r = k >> 6, c = k & 63;
    WlBT[c * 128 + r] = (__bf16)Wl[k];
  }
}

// ---------------- generic GEMM: C[M x 128] = A(f32,[M x K]) @ B, B given transposed ----------------
// BT is bf16 [128][K]. Block = 256 threads = 8 waves, 128x128 tile; wave grid 4(M) x 2(N).
__global__ void gemm_f32xbf16_n128(const float* __restrict__ A, const __bf16* __restrict__ BT,
                                   float* __restrict__ C, int K, int lda) {
  __shared__ __align__(16) __bf16 As[128 * LDT];
  __shared__ __align__(16) __bf16 Bs[128 * LDT];
  const int t = threadIdx.x;
  const int lane = t & 31;
  const int wid = t >> 5;
  const int wm = wid >> 1;  // 0..3
  const int wn = wid & 1;   // 0..1
  const int m0 = blockIdx.x * 128;

  v8f acc[2][4] = {};

  for (int kc = 0; kc < K; kc += 32) {
    // Stage A tile (128 rows x 32 k, f32 -> bf16): 1024 float4, 4 per thread.
#pragma unroll
    for (int it = 0; it < 4; ++it) {
      const int i = t + 256 * it;
      const int r = i >> 3, c4 = i & 7;
      const float4 v = *reinterpret_cast<const float4*>(
          A + (size_t)(m0 + r) * lda + kc + c4 * 4);
      const v4bf pk = {(__bf16)v.x, (__bf16)v.y, (__bf16)v.z, (__bf16)v.w};
      *reinterpret_cast<v4bf*>(As + r * LDT + c4 * 4) = pk;
    }
    // Stage B tile (128 cols x 32 k bf16): 512 uint4, 2 per thread.
#pragma unroll
    for (int it = 0; it < 2; ++it) {
      const int i = t + 256 * it;
      const int r = i >> 2, c = i & 3;
      *reinterpret_cast<uint4*>(Bs + r * LDT + c * 8) =
          *reinterpret_cast<const uint4*>(BT + (size_t)r * K + kc + c * 8);
    }
    if (kc + 32 < K)  // hint next A tile into cache (global_prefetch_b8)
      __builtin_prefetch(A + (size_t)(m0 + (t >> 1)) * lda + kc + 32, 0, 1);
    __syncthreads();

    v16bf af[2], bfr[4];
#pragma unroll
    for (int ms = 0; ms < 2; ++ms) af[ms] = load_frag(As, LDT, wm * 32 + ms * 16, 0, lane);
#pragma unroll
    for (int ns = 0; ns < 4; ++ns) bfr[ns] = load_frag(Bs, LDT, wn * 64 + ns * 16, 0, lane);
#pragma unroll
    for (int ms = 0; ms < 2; ++ms)
#pragma unroll
      for (int ns = 0; ns < 4; ++ns)
        acc[ms][ns] = wmma_bf16(af[ms], bfr[ns], acc[ms][ns]);
    __syncthreads();
  }

  // D layout: VGPR r -> row r (+8 for lanes 16..31), col = lane&15.
  const int rofs = (lane >> 4) << 3;
#pragma unroll
  for (int ms = 0; ms < 2; ++ms)
#pragma unroll
    for (int ns = 0; ns < 4; ++ns) {
      const int col = wn * 64 + ns * 16 + (lane & 15);
#pragma unroll
      for (int r = 0; r < 8; ++r) {
        const int row = m0 + wm * 32 + ms * 16 + r + rofs;
        C[(size_t)row * 128 + col] = acc[ms][ns][r];
      }
    }
}

// ---------------- row dots: hs = h . a_src, hd = h . a_dst (wave per row) ----------------
__global__ void rowdot_kernel(const float* __restrict__ h, const float* __restrict__ a_src,
                              const float* __restrict__ a_dst, float* hs, float* hd, int N) {
  const int row = blockIdx.x * 8 + (threadIdx.x >> 5);
  if (row >= N) return;
  const int lane = threadIdx.x & 31;
  const float4 hv = *reinterpret_cast<const float4*>(h + (size_t)row * 128 + lane * 4);
  const float4 as = *reinterpret_cast<const float4*>(a_src + lane * 4);
  const float4 ad = *reinterpret_cast<const float4*>(a_dst + lane * 4);
  float s1 = hv.x * as.x + hv.y * as.y + hv.z * as.z + hv.w * as.w;
  float s2 = hv.x * ad.x + hv.y * ad.y + hv.z * ad.z + hv.w * ad.w;
#pragma unroll
  for (int off = 16; off > 0; off >>= 1) {
    s1 += __shfl_xor(s1, off, 32);
    s2 += __shfl_xor(s2, off, 32);
  }
  if (lane == 0) { hs[row] = s1; hd[row] = s2; }
}

// ---------------- edge passes ----------------
__device__ inline void atomicMaxF(float* addr, float val) {
  // sign-split trick: int-max for non-negative, uint-min for negative
  if (!(val < 0.0f)) atomicMax(reinterpret_cast<int*>(addr), __float_as_int(val));
  else               atomicMin(reinterpret_cast<unsigned int*>(addr), __float_as_uint(val));
}

__device__ inline void edge_sd(const int* ei, int e, int E, int& s, int& d) {
  if (e < E) { s = ei[e]; d = ei[E + e]; } else { s = d = e - E; }
}

__global__ void edge_max_kernel(const int* __restrict__ ei, const float* __restrict__ hs,
                                const float* __restrict__ hd, float* m, int E, int total) {
  const int e = blockIdx.x * blockDim.x + threadIdx.x;
  if (e >= total) return;
  int s, d; edge_sd(ei, e, E, s, d);
  float a = hs[s] + hd[d];
  a = (a >= 0.0f) ? a : 0.2f * a;  // leaky_relu
  atomicMaxF(m + d, a);
}

__global__ void edge_exp_kernel(const int* __restrict__ ei, const float* __restrict__ hs,
                                const float* __restrict__ hd, const float* __restrict__ m,
                                float* __restrict__ ebuf, float* denom, int E, int total) {
  const int e = blockIdx.x * blockDim.x + threadIdx.x;
  if (e >= total) return;
  int s, d; edge_sd(ei, e, E, s, d);
  float a = hs[s] + hd[d];
  a = (a >= 0.0f) ? a : 0.2f * a;
  const float ev = expf(a - m[d]);
  ebuf[e] = ev;
  atomicAdd(denom + d, ev);
}

// wave per edge: gat[d,:] += w * h[s,:]  (128 feats = 32 lanes x float4)
__global__ void edge_scatter_kernel(const int* __restrict__ ei, const float* __restrict__ ebuf,
                                    const float* __restrict__ denom, const float* __restrict__ h,
                                    float* __restrict__ gat, int E, int total) {
  const int e = blockIdx.x * 8 + (threadIdx.x >> 5);
  if (e >= total) return;
  const int lane = threadIdx.x & 31;
  int s, d; edge_sd(ei, e, E, s, d);
  const float w = ebuf[e] / denom[d];
  const float4 hv = *reinterpret_cast<const float4*>(h + (size_t)s * 128 + lane * 4);
  float* gp = gat + (size_t)d * 128 + lane * 4;
  atomicAdd(gp + 0, hv.x * w);
  atomicAdd(gp + 1, hv.y * w);
  atomicAdd(gp + 2, hv.z * w);
  atomicAdd(gp + 3, hv.w * w);
}

// gat += b_gat, convert to bf16 *transposed* ([128][Nitems]) via LDS tile transpose.
// Block handles a 128-row x 128-col tile; both global phases fully coalesced.
__global__ void biascvt_T_kernel(const float* __restrict__ gat, const float* __restrict__ b,
                                 __bf16* __restrict__ gatBT, int Nitems) {
  __shared__ float tile[128][129];
  const int t = threadIdx.x;
  const int r0 = blockIdx.x * 128;
#pragma unroll
  for (int it = 0; it < 16; ++it) {
    const int i = t + 256 * it;          // 0..4095 float4 slots
    const int r = i >> 5, c4 = i & 31;
    const float4 v  = *reinterpret_cast<const float4*>(gat + (size_t)(r0 + r) * 128 + c4 * 4);
    const float4 bv = *reinterpret_cast<const float4*>(b + c4 * 4);
    tile[r][c4 * 4 + 0] = v.x + bv.x;
    tile[r][c4 * 4 + 1] = v.y + bv.y;
    tile[r][c4 * 4 + 2] = v.z + bv.z;
    tile[r][c4 * 4 + 3] = v.w + bv.w;
  }
  __syncthreads();
#pragma unroll
  for (int it = 0; it < 16; ++it) {
    const int i = t + 256 * it;          // 0..4095 v4bf slots
    const int col = i >> 5, k4 = i & 31;
    const v4bf pk = {(__bf16)tile[k4 * 4 + 0][col], (__bf16)tile[k4 * 4 + 1][col],
                     (__bf16)tile[k4 * 4 + 2][col], (__bf16)tile[k4 * 4 + 3][col]};
    *reinterpret_cast<v4bf*>(gatBT + (size_t)col * Nitems + r0 + k4 * 4) = pk;
  }
}

// ---------------- fused tail: out = leaky(C1 @ W_etn + b_etn) @ W_lin ----------------
__global__ void tail_kernel(const float* __restrict__ C1, const __bf16* __restrict__ WeBT,
                            const float* __restrict__ b_etn, const __bf16* __restrict__ WlBT,
                            float* __restrict__ out) {
  __shared__ __align__(16) __bf16 As[128 * LDT];
  __shared__ __align__(16) __bf16 Bs[128 * LDT];
  __shared__ __align__(16) __bf16 Ts[128 * LDW];  // stage-1 result, row-major [row][k]
  __shared__ __align__(16) __bf16 Wl[64 * LDW];   // W_lin^T, [n][k]
  const int t = threadIdx.x;
  const int lane = t & 31;
  const int wid = t >> 5;
  const int wm = wid >> 1, wn = wid & 1;
  const int m0 = blockIdx.x * 128;
  const int rofs = (lane >> 4) << 3;

  // preload W_lin^T (64 x 128 bf16): 1024 uint4
#pragma unroll
  for (int it = 0; it < 4; ++it) {
    const int i = t + 256 * it;
    const int r = i >> 4, c = i & 15;
    *reinterpret_cast<uint4*>(Wl + r * LDW + c * 8) =
        *reinterpret_cast<const uint4*>(WlBT + (size_t)r * 128 + c * 8);
  }

  // Stage 1: T = leaky(C1_tile @ W_etn + b_etn)  (K = 128)
  v8f acc[2][4] = {};
  for (int kc = 0; kc < 128; kc += 32) {
#pragma unroll
    for (int it = 0; it < 4; ++it) {
      const int i = t + 256 * it;
      const int r = i >> 3, c4 = i & 7;
      const float4 v = *reinterpret_cast<const float4*>(
          C1 + (size_t)(m0 + r) * 128 + kc + c4 * 4);
      const v4bf pk = {(__bf16)v.x, (__bf16)v.y, (__bf16)v.z, (__bf16)v.w};
      *reinterpret_cast<v4bf*>(As + r * LDT + c4 * 4) = pk;
    }
#pragma unroll
    for (int it = 0; it < 2; ++it) {
      const int i = t + 256 * it;
      const int r = i >> 2, c = i & 3;
      *reinterpret_cast<uint4*>(Bs + r * LDT + c * 8) =
          *reinterpret_cast<const uint4*>(WeBT + (size_t)r * 128 + kc + c * 8);
    }
    __syncthreads();
    v16bf af[2], bfr[4];
#pragma unroll
    for (int ms = 0; ms < 2; ++ms) af[ms] = load_frag(As, LDT, wm * 32 + ms * 16, 0, lane);
#pragma unroll
    for (int ns = 0; ns < 4; ++ns) bfr[ns] = load_frag(Bs, LDT, wn * 64 + ns * 16, 0, lane);
#pragma unroll
    for (int ms = 0; ms < 2; ++ms)
#pragma unroll
      for (int ns = 0; ns < 4; ++ns)
        acc[ms][ns] = wmma_bf16(af[ms], bfr[ns], acc[ms][ns]);
    __syncthreads();
  }

  // epilogue: bias + leaky -> Ts (bf16 row-major, serves as A of stage 2)
#pragma unroll
  for (int ms = 0; ms < 2; ++ms)
#pragma unroll
    for (int ns = 0; ns < 4; ++ns) {
      const int col = wn * 64 + ns * 16 + (lane & 15);
      const float bias = b_etn[col];
#pragma unroll
      for (int r = 0; r < 8; ++r) {
        const int row = wm * 32 + ms * 16 + r + rofs;
        float v = acc[ms][ns][r] + bias;
        v = (v >= 0.0f) ? v : 0.2f * v;
        Ts[row * LDW + col] = (__bf16)v;
      }
    }
  __syncthreads();

  // Stage 2: out_tile(128x64) = Ts @ W_lin  (K = 128; wave grid 4M x 2N, 32x32 per wave)
  v8f acc2[2][2] = {};
#pragma unroll
  for (int kc = 0; kc < 128; kc += 32) {
    v16bf af[2], bfr[2];
#pragma unroll
    for (int ms = 0; ms < 2; ++ms) af[ms] = load_frag(Ts, LDW, wm * 32 + ms * 16, kc, lane);
#pragma unroll
    for (int ns = 0; ns < 2; ++ns) bfr[ns] = load_frag(Wl, LDW, wn * 32 + ns * 16, kc, lane);
#pragma unroll
    for (int ms = 0; ms < 2; ++ms)
#pragma unroll
      for (int ns = 0; ns < 2; ++ns)
        acc2[ms][ns] = wmma_bf16(af[ms], bfr[ns], acc2[ms][ns]);
  }
#pragma unroll
  for (int ms = 0; ms < 2; ++ms)
#pragma unroll
    for (int ns = 0; ns < 2; ++ns) {
      const int col = wn * 32 + ns * 16 + (lane & 15);
#pragma unroll
      for (int r = 0; r < 8; ++r) {
        const int row = m0 + wm * 32 + ms * 16 + r + rofs;
        out[(size_t)row * 64 + col] = acc2[ms][ns][r];
      }
    }
}

// ---------------- launch ----------------
extern "C" void kernel_launch(void* const* d_in, const int* in_sizes, int n_in,
                              void* d_out, int out_size, void* d_ws, size_t ws_size,
                              hipStream_t stream) {
  const float* x     = (const float*)d_in[0];
  const float* Hm    = (const float*)d_in[1];
  const int*   ei    = (const int*)d_in[2];
  const float* Wg    = (const float*)d_in[3];
  const float* a_src = (const float*)d_in[4];
  const float* a_dst = (const float*)d_in[5];
  const float* b_gat = (const float*)d_in[6];
  const float* We    = (const float*)d_in[7];
  const float* b_etn = (const float*)d_in[8];
  const float* Wl    = (const float*)d_in[9];

  const int Fin = 128, Hd = 128, Do = 64;
  const int Nitems = in_sizes[0] / Fin;       // 16384
  const int Nnodes = in_sizes[1] / Nitems;    // 8192
  const int E      = in_sizes[2] / 2;         // 524288
  const int total  = E + Nitems;

  char* w = (char*)d_ws;
  auto carve = [&](size_t bytes) -> char* {
    char* p = w;
    w += (bytes + 255) & ~(size_t)255;
    return p;
  };
  float*  h     = (float*)carve((size_t)Nitems * Hd * 4);
  float*  gat   = (float*)carve((size_t)Nitems * Hd * 4);
  float*  hs    = (float*)carve((size_t)Nitems * 4);
  float*  hd    = (float*)carve((size_t)Nitems * 4);
  float*  mmax  = (float*)carve((size_t)Nitems * 4);
  float*  denom = (float*)carve((size_t)Nitems * 4);
  float*  ebuf  = (float*)carve((size_t)total * 4);
  __bf16* gatBT = (__bf16*)carve((size_t)Nitems * Hd * 2);  // [128][Nitems]
  float*  C1    = (float*)carve((size_t)Nnodes * Hd * 4);
  __bf16* WgBT  = (__bf16*)carve((size_t)Fin * Hd * 2);     // [128][128]
  __bf16* WeBT  = (__bf16*)carve((size_t)Hd * Hd * 2);      // [128][128]
  __bf16* WlBT  = (__bf16*)carve((size_t)Hd * Do * 2);      // [64][128]

  init_kernel<<<2048, 256, 0, stream>>>(mmax, denom, gat, Nitems, Nitems * Hd,
                                        Wg, WgBT, We, WeBT, Fin * Hd, Wl, WlBT, Hd * Do);

  // h = x @ W_gat
  gemm_f32xbf16_n128<<<Nitems / 128, 256, 0, stream>>>(x, WgBT, h, Fin, Fin);

  rowdot_kernel<<<Nitems / 8, 256, 0, stream>>>(h, a_src, a_dst, hs, hd, Nitems);

  edge_max_kernel<<<(total + 255) / 256, 256, 0, stream>>>(ei, hs, hd, mmax, E, total);
  edge_exp_kernel<<<(total + 255) / 256, 256, 0, stream>>>(ei, hs, hd, mmax, ebuf, denom, E, total);
  edge_scatter_kernel<<<(total + 7) / 8, 256, 0, stream>>>(ei, ebuf, denom, h, gat, E, total);

  biascvt_T_kernel<<<Nitems / 128, 256, 0, stream>>>(gat, b_gat, gatBT, Nitems);

  // C1 = H @ gat_out   (HBM-bound: streams 512 MB of H once)
  gemm_f32xbf16_n128<<<Nnodes / 128, 256, 0, stream>>>(Hm, gatBT, C1, Nitems, Nitems);

  // out = leaky(C1 @ W_etn + b_etn) @ W_lin
  tail_kernel<<<Nnodes / 128, 256, 0, stream>>>(C1, WeBT, b_etn, WlBT, (float*)d_out);
}
